// Nequix_26156350832704
// MI455X (gfx1250) — matrix operand order repaired and checked by reading, not compiled, not run on hardware
//
#include <hip/hip_runtime.h>
#include <math.h>

#define N_NODES 10000
#define N_EDGES 100000
#define N_SPEC  5
#define HID     128

typedef __attribute__((ext_vector_type(16))) _Float16 v16h;
typedef __attribute__((ext_vector_type(8)))  float    v8f;

__device__ __forceinline__ float silu_f(float x) {
    return x * __builtin_amdgcn_rcpf(1.0f + __expf(-x));
}

// ---------------------------------------------------------------------------
// Weight pre-pack: W (S,K,N) f32 row-major -> zero-padded f16 fragments in the
// contiguous-k wave32 WMMA B order this file uses (matches the A loads). Per (ktile,ntile): 32 lanes x 16 halves where
// K = (lane>>4)*16 + h, col = lane&15. Kp=ceil32(K), Np=ceil16(N).
// ---------------------------------------------------------------------------
__global__ void pack_w(const float* __restrict__ W, _Float16* __restrict__ out,
                       int K, int N, int Kp, int Np, int S)
{
    long idx = (long)blockIdx.x * blockDim.x + threadIdx.x;
    long per = (long)Kp * Np;
    if (idx >= per * S) return;
    int  s   = (int)(idx / per);
    long r   = idx - (long)s * per;
    int  ntiles = Np >> 4;
    int  frag = (int)(r >> 9);
    int  rem  = (int)(r & 511);
    int  lane = rem >> 4, h = rem & 15;
    int  kt = frag / ntiles, nt = frag - kt * ntiles;
    int  k = kt * 32 + ((lane >> 4) << 4) + h;
    int  n = nt * 16 + (lane & 15);
    float v = (k < K && n < N) ? W[(size_t)s * K * N + (size_t)k * N + n] : 0.0f;
    out[idx] = (_Float16)v;
}

// ---------------------------------------------------------------------------
// Register-tiled GEMM: each wave computes a 16 x (NT*16) strip; A fragments
// loaded once per K-chunk and reused across NT WMMAs. B fragments staged in
// LDS once per block (8-wave + NT reuse). K compile-time, M % 16 == 0,
// Np % (NT*16) == 0.
// ---------------------------------------------------------------------------
template<int K, int NT, int ACC, int ACT>
__global__ __launch_bounds__(256)
void gemm_k(const float* __restrict__ A, const _Float16* __restrict__ Wpk,
            float* __restrict__ C, int M, int ntiles, int ldC, float alpha)
{
    __shared__ __align__(32) unsigned int bsh[(K / 32) * NT * 256];
    const int nt0 = blockIdx.x * NT;
    const unsigned int* wsrc = (const unsigned int*)Wpk;
#pragma unroll
    for (int kc = 0; kc < K / 32; ++kc)
#pragma unroll
        for (int j = 0; j < NT; ++j)
            bsh[(kc * NT + j) * 256 + threadIdx.x] =
                wsrc[(size_t)(kc * ntiles + nt0 + j) * 256 + threadIdx.x];
    __syncthreads();

    const int lane = threadIdx.x & 31;
    const int wave = threadIdx.x >> 5;
    const int row0 = (blockIdx.y * 8 + wave) * 16;
    if (row0 >= M) return;                        // wave-uniform, after barrier

    const float* ap = A + (size_t)(row0 + (lane & 15)) * K + (lane >> 4) * 16;
    const _Float16* bl = (const _Float16*)bsh + lane * 16;

    v8f c[NT];
#pragma unroll
    for (int j = 0; j < NT; ++j) c[j] = (v8f){};

#pragma unroll
    for (int kc = 0; kc < K / 32; ++kc) {
        const float4 a0 = *(const float4*)(ap + kc * 32);
        const float4 a1 = *(const float4*)(ap + kc * 32 + 4);
        const float4 a2 = *(const float4*)(ap + kc * 32 + 8);
        const float4 a3 = *(const float4*)(ap + kc * 32 + 12);
        v16h a;
        a[0]=(_Float16)a0.x;  a[1]=(_Float16)a0.y;  a[2]=(_Float16)a0.z;  a[3]=(_Float16)a0.w;
        a[4]=(_Float16)a1.x;  a[5]=(_Float16)a1.y;  a[6]=(_Float16)a1.z;  a[7]=(_Float16)a1.w;
        a[8]=(_Float16)a2.x;  a[9]=(_Float16)a2.y;  a[10]=(_Float16)a2.z; a[11]=(_Float16)a2.w;
        a[12]=(_Float16)a3.x; a[13]=(_Float16)a3.y; a[14]=(_Float16)a3.z; a[15]=(_Float16)a3.w;
#pragma unroll
        for (int j = 0; j < NT; ++j) {
            const v16h b = *(const v16h*)(bl + (kc * NT + j) * 512);
            c[j] = __builtin_amdgcn_wmma_f32_16x16x32_f16(false, a, false, b,
                                                          (short)0, c[j], false, false);
        }
    }
    const int crow = row0 + (lane >> 4) * 8;
#pragma unroll
    for (int j = 0; j < NT; ++j) {
        float* cp = C + (size_t)crow * ldC + (nt0 + j) * 16 + (lane & 15);
        float cold[8];
        if constexpr (ACC) {
#pragma unroll
            for (int i = 0; i < 8; ++i) cold[i] = cp[(size_t)i * ldC];
        }
#pragma unroll
        for (int i = 0; i < 8; ++i) {
            float v = alpha * c[j][i];
            if constexpr (ACC) v += cold[i];
            if constexpr (ACT) v = silu_f(v);
            cp[(size_t)i * ldC] = v;
        }
    }
}

// ---------------------------------------------------------------------------
// Species-gathered skip GEMM: C[r,:] (+)= alpha * A[r,:] @ Wt[species[r]].
// NT=2 register tiling; all 5 species' fragments staged in LDS; A fragments
// loaded once; per species scaled by 0/1 half (packed f16 muls) so EXEC
// stays all-ones for every WMMA.
// ---------------------------------------------------------------------------
template<int K, int NT, int ACC, int ACT>
__global__ __launch_bounds__(256)
void gemm_skip_k(const float* __restrict__ A, const _Float16* __restrict__ Wpk,
                 const int* __restrict__ spec, float* __restrict__ C,
                 int M, int ntiles, int ldC, float alpha)
{
    __shared__ __align__(32) unsigned int bsh[N_SPEC * (K / 32) * NT * 256];
    const int nt0 = blockIdx.x * NT;
    const unsigned int* wsrc = (const unsigned int*)Wpk;
    const size_t sliceU = (size_t)(K / 32) * ntiles * 256;
#pragma unroll
    for (int t = 0; t < N_SPEC; ++t)
#pragma unroll
        for (int kc = 0; kc < K / 32; ++kc)
#pragma unroll
            for (int j = 0; j < NT; ++j)
                bsh[((t * (K / 32) + kc) * NT + j) * 256 + threadIdx.x] =
                    wsrc[(size_t)t * sliceU
                         + (size_t)(kc * ntiles + nt0 + j) * 256 + threadIdx.x];
    __syncthreads();

    const int lane = threadIdx.x & 31;
    const int wave = threadIdx.x >> 5;
    const int row0 = (blockIdx.y * 8 + wave) * 16;
    if (row0 >= M) return;

    const int myspec = spec[row0 + (lane & 15)];
    const float* ap = A + (size_t)(row0 + (lane & 15)) * K + (lane >> 4) * 16;

    v16h af[K / 32];
#pragma unroll
    for (int kc = 0; kc < K / 32; ++kc) {
        const float4 a0 = *(const float4*)(ap + kc * 32);
        const float4 a1 = *(const float4*)(ap + kc * 32 + 4);
        const float4 a2 = *(const float4*)(ap + kc * 32 + 8);
        const float4 a3 = *(const float4*)(ap + kc * 32 + 12);
        v16h a;
        a[0]=(_Float16)a0.x;  a[1]=(_Float16)a0.y;  a[2]=(_Float16)a0.z;  a[3]=(_Float16)a0.w;
        a[4]=(_Float16)a1.x;  a[5]=(_Float16)a1.y;  a[6]=(_Float16)a1.z;  a[7]=(_Float16)a1.w;
        a[8]=(_Float16)a2.x;  a[9]=(_Float16)a2.y;  a[10]=(_Float16)a2.z; a[11]=(_Float16)a2.w;
        a[12]=(_Float16)a3.x; a[13]=(_Float16)a3.y; a[14]=(_Float16)a3.z; a[15]=(_Float16)a3.w;
        af[kc] = a;
    }

    const _Float16* bl = (const _Float16*)bsh + lane * 16;
    v8f c[NT];
#pragma unroll
    for (int j = 0; j < NT; ++j) c[j] = (v8f){};

    for (int t = 0; t < N_SPEC; ++t) {
        const _Float16 hm = (myspec == t) ? (_Float16)1.0f : (_Float16)0.0f;
#pragma unroll
        for (int kc = 0; kc < K / 32; ++kc) {
            const v16h am = af[kc] * hm;
#pragma unroll
            for (int j = 0; j < NT; ++j) {
                const v16h b = *(const v16h*)(bl + ((t * (K / 32) + kc) * NT + j) * 512);
                c[j] = __builtin_amdgcn_wmma_f32_16x16x32_f16(false, am, false, b,
                                                              (short)0, c[j], false, false);
            }
        }
    }
    const int crow = row0 + (lane >> 4) * 8;
#pragma unroll
    for (int j = 0; j < NT; ++j) {
        float* cp = C + (size_t)crow * ldC + (nt0 + j) * 16 + (lane & 15);
        float cold[8];
        if constexpr (ACC) {
#pragma unroll
            for (int i = 0; i < 8; ++i) cold[i] = cp[(size_t)i * ldC];
        }
#pragma unroll
        for (int i = 0; i < 8; ++i) {
            float v = alpha * c[j][i];
            if constexpr (ACC) v += cold[i];
            if constexpr (ACT) v = silu_f(v);
            cp[(size_t)i * ldC] = v;
        }
    }
}

// ---------------------------------------------------------------------------
__global__ void edge_geom(const float* __restrict__ disp,
                          float* __restrict__ rb, float* __restrict__ shv)
{
    int e = blockIdx.x * blockDim.x + threadIdx.x;
    if (e >= N_EDGES) return;
    float dx = disp[3 * e], dy = disp[3 * e + 1], dz = disp[3 * e + 2];
    float r2 = dx * dx + dy * dy + dz * dz;
    float r  = (r2 == 0.0f) ? 0.0f : __fsqrt_rn(r2);
    const float rmax = 5.0f;
    float u  = r * (1.0f / rmax);
    float u2 = u * u;
    float cut = 1.0f - 6.0f * u2 + 8.0f * u2 * u - 3.0f * u2 * u2;
    cut = (u < 1.0f) ? cut : 0.0f;
    float invr = (r == 0.0f) ? 1.0f : (1.0f / r);
    const float pref = 2.0f / rmax;
#pragma unroll
    for (int k = 0; k < 8; ++k) {
        float w = (float)(k + 1) * 3.14159265358979f;
        float b = (r == 0.0f) ? (w / rmax) : (__sinf(w * r / rmax) * invr);
        rb[(size_t)e * 32 + k] = pref * b * cut;
    }
#pragma unroll
    for (int k = 8; k < 32; ++k) rb[(size_t)e * 32 + k] = 0.0f;
    const float s3 = 1.7320508075688772f;
    shv[3 * e]     = s3 * dx * invr;
    shv[3 * e + 1] = s3 * dy * invr;
    shv[3 * e + 2] = s3 * dz * invr;
}

__global__ void onehot_k(const int* __restrict__ spec, float* __restrict__ s)
{
    int n = blockIdx.x * blockDim.x + threadIdx.x;
    if (n >= N_NODES) return;
    int t = spec[n];
#pragma unroll
    for (int i = 0; i < 32; ++i)
        s[(size_t)n * 32 + i] = (i == t) ? 1.0f : 0.0f;
}

// ---------------------------------------------------------------------------
__global__ void tp_scatter(const float* __restrict__ s1, const float* __restrict__ v1,
                           const float* __restrict__ rbuf, const float* __restrict__ shv,
                           const int* __restrict__ snd, const int* __restrict__ rcv,
                           float* __restrict__ agg0, float* __restrict__ agg1,
                           int n0_in, int n1_in, int n0_tp, int n1_tp,
                           int s1_ld, int v1_ld, int r_ld, int a0_ld, int a1_ld)
{
    const int tot = n0_tp + n1_tp;
    long idx = (long)blockIdx.x * blockDim.x + threadIdx.x;
    if (idx >= (long)N_EDGES * tot) return;
    int e    = (int)(idx / tot);
    int cidx = (int)(idx - (long)e * tot);
    int sN = snd[e], rN = rcv[e];
    float w = rbuf[(size_t)e * r_ld + cidx] * 0.31622776601683794f;
    float sh0 = shv[3 * e], sh1 = shv[3 * e + 1], sh2 = shv[3 * e + 2];

    if (cidx < n0_tp) {
        float t0;
        if (cidx < n0_in) {
            t0 = s1[(size_t)sN * s1_ld + cidx];
        } else {
            int i = cidx - n0_in;
            size_t ps = (size_t)N_NODES * v1_ld;
            t0 = (v1[(size_t)sN * v1_ld + i] * sh0 +
                  v1[ps + (size_t)sN * v1_ld + i] * sh1 +
                  v1[2 * ps + (size_t)sN * v1_ld + i] * sh2) * 0.5773502691896258f;
        }
        atomicAdd(&agg0[(size_t)rN * a0_ld + cidx], t0 * w);
    } else {
        int j = cidx - n0_tp;
        size_t pd = (size_t)N_NODES * a1_ld;
        float t0, t1c, t2;
        if (j < n0_in) {
            float sv = s1[(size_t)sN * s1_ld + j];
            t0 = sv * sh0; t1c = sv * sh1; t2 = sv * sh2;
        } else {
            int i = j - n0_in;
            size_t ps = (size_t)N_NODES * v1_ld;
            t0  = v1[(size_t)sN * v1_ld + i];
            t1c = v1[ps + (size_t)sN * v1_ld + i];
            t2  = v1[2 * ps + (size_t)sN * v1_ld + i];
        }
        atomicAdd(&agg1[(size_t)rN * a1_ld + j], t0 * w);
        atomicAdd(&agg1[pd + (size_t)rN * a1_ld + j], t1c * w);
        atomicAdd(&agg1[2 * pd + (size_t)rN * a1_ld + j], t2 * w);
    }
}

__global__ void gate_k(const float* __restrict__ out0, const float* __restrict__ out1,
                       float* __restrict__ s_new, float* __restrict__ v_new, int n0_out)
{
    int idx = blockIdx.x * blockDim.x + threadIdx.x;
    if (idx >= N_NODES * HID) return;
    int n = idx / HID, i = idx % HID;
    int H = n0_out - HID;
    float sv = silu_f(out0[(size_t)n * n0_out + i]);
    float g  = silu_f(out0[(size_t)n * n0_out + H + i]);
    s_new[(size_t)n * HID + i] = sv;
    size_t pd = (size_t)N_NODES * HID;
    v_new[(size_t)n * HID + i]          = out1[(size_t)n * HID + i] * g;
    v_new[pd + (size_t)n * HID + i]     = out1[pd + (size_t)n * HID + i] * g;
    v_new[2 * pd + (size_t)n * HID + i] = out1[2 * pd + (size_t)n * HID + i] * g;
}

__global__ void readout_k(const float* __restrict__ s, const float* __restrict__ w,
                          const float* __restrict__ atom_e, const int* __restrict__ spec,
                          float* __restrict__ out)
{
    int n = blockIdx.x * blockDim.x + threadIdx.x;
    if (n >= N_NODES) return;
    float acc = 0.0f;
    for (int i = 0; i < HID; ++i)
        acc += s[(size_t)n * HID + i] * w[i];
    out[n] = acc * 0.08838834764831845f + atom_e[spec[n]];
}

// ---------------------------------------------------------------------------
extern "C" void kernel_launch(void* const* d_in, const int* in_sizes, int n_in,
                              void* d_out, int out_size, void* d_ws, size_t ws_size,
                              hipStream_t stream)
{
    (void)in_sizes; (void)n_in; (void)out_size; (void)ws_size;
    const float* disp      = (const float*)d_in[0];
    const int*   species   = (const int*)d_in[1];
    const int*   senders   = (const int*)d_in[2];
    const int*   receivers = (const int*)d_in[3];
    const float* atom_e    = (const float*)d_in[4];
    const float* L0_lin1_0 = (const float*)d_in[5];
    const float* L0_lin2_0 = (const float*)d_in[6];
    const float* L0_lin2_1 = (const float*)d_in[7];
    const float* L0_r[4]   = {(const float*)d_in[8],(const float*)d_in[9],
                              (const float*)d_in[10],(const float*)d_in[11]};
    const float* L0_skip0  = (const float*)d_in[12];
    const float* L1_lin1_0 = (const float*)d_in[13];
    const float* L1_lin1_1 = (const float*)d_in[14];
    const float* L1_lin2_0 = (const float*)d_in[15];
    const float* L1_lin2_1 = (const float*)d_in[16];
    const float* L1_r[4]   = {(const float*)d_in[17],(const float*)d_in[18],
                              (const float*)d_in[19],(const float*)d_in[20]};
    const float* L1_skip0  = (const float*)d_in[21];
    const float* L1_skip1  = (const float*)d_in[22];
    const float* L2_lin1_0 = (const float*)d_in[23];
    const float* L2_lin1_1 = (const float*)d_in[24];
    const float* L2_lin2_0 = (const float*)d_in[25];
    const float* L2_r[4]   = {(const float*)d_in[26],(const float*)d_in[27],
                              (const float*)d_in[28],(const float*)d_in[29]};
    const float* L2_skip0  = (const float*)d_in[30];
    const float* readout   = (const float*)d_in[31];
    float* out = (float*)d_out;

    float* ws = (float*)d_ws;
    size_t cur = 0;
    auto alloc = [&](size_t n) { float* p = ws + cur; cur += n; return p; };
    float* rb    = alloc((size_t)N_EDGES * 32);
    float* shv   = alloc((size_t)N_EDGES * 3 + 16);
    float* rbuf  = alloc((size_t)N_EDGES * 512);
    float* h1    = alloc((size_t)N_EDGES * 64);
    float* h2    = alloc((size_t)N_EDGES * 64);
    float* s_buf = alloc((size_t)N_NODES * 128);
    float* s1b   = alloc((size_t)N_NODES * 128);
    float* v_buf = alloc((size_t)3 * N_NODES * 128);
    float* v1b   = alloc((size_t)3 * N_NODES * 128);
    float* agg0  = alloc((size_t)N_NODES * 256);
    float* agg1  = alloc((size_t)3 * N_NODES * 256);
    float* out0  = alloc((size_t)N_NODES * 256);
    float* out1  = alloc((size_t)3 * N_NODES * 128);

    _Float16* pkbase = (_Float16*)alloc(600000);
    size_t pkcur = 0;
    auto pack = [&](const float* W, int K, int N, int S) -> _Float16* {
        int Kp = (K + 31) & ~31, Np = (N + 15) & ~15;
        size_t total = (size_t)S * Kp * Np;
        _Float16* o = pkbase + pkcur;
        pkcur += total;
        pack_w<<<(unsigned)((total + 255) / 256), 256, 0, stream>>>(W, o, K, N, Kp, Np, S);
        return o;
    };

    const float is5   = 0.4472135954999579f;
    const float is128 = 0.08838834764831845f;
    const float is256 = 0.0625f;

    auto gemm = [&](int K, int NT, const float* A, const _Float16* Wpk, float* C,
                    int M, int Np, int ldC, float al, int act) {
        dim3 grid(Np / (NT * 16), (M / 16 + 7) / 8);
        int ntl = Np / 16;
#define GK(KK,NN) do { \
        if (act) gemm_k<KK,NN,0,1><<<grid,256,0,stream>>>(A,Wpk,C,M,ntl,ldC,al); \
        else     gemm_k<KK,NN,0,0><<<grid,256,0,stream>>>(A,Wpk,C,M,ntl,ldC,al); } while(0)
        if      (K == 32  && NT == 1) GK(32,1);
        else if (K == 32  && NT == 4) GK(32,4);
        else if (K == 32  && NT == 8) GK(32,8);
        else if (K == 64  && NT == 1) GK(64,1);
        else if (K == 64  && NT == 4) GK(64,4);
        else if (K == 64  && NT == 8) GK(64,8);
        else if (K == 128 && NT == 8) GK(128,8);
        else if (K == 256 && NT == 8) GK(256,8);
#undef GK
    };
    auto gskip = [&](int K, const float* A, const _Float16* Wpk, float* C,
                     int M, int Np, int ldC, float al, int act) {
        dim3 grid(Np / 32, (M / 16 + 7) / 8);     // NT = 2
        int ntl = Np / 16;
#define GS(KK) do { \
        if (act) gemm_skip_k<KK,2,1,1><<<grid,256,0,stream>>>(A,Wpk,species,C,M,ntl,ldC,al); \
        else     gemm_skip_k<KK,2,1,0><<<grid,256,0,stream>>>(A,Wpk,species,C,M,ntl,ldC,al); } while(0)
        switch (K) {
        case 32:  GS(32);  break;
        case 128: GS(128); break;
        }
#undef GS
    };

    _Float16* p0r0 = pack(L0_r[0], 8, 64, 1);
    _Float16* p0r1 = pack(L0_r[1], 64, 64, 1);
    _Float16* p0r2 = pack(L0_r[2], 64, 64, 1);
    _Float16* p0r3 = pack(L0_r[3], 64, 10, 1);
    _Float16* p0l1 = pack(L0_lin1_0, 5, 5, 1);
    _Float16* p0l20= pack(L0_lin2_0, 5, 256, 1);
    _Float16* p0l21= pack(L0_lin2_1, 5, 128, 1);
    _Float16* p0sk = pack(L0_skip0, 5, 256, N_SPEC);
    _Float16* p1r0 = pack(L1_r[0], 8, 64, 1);
    _Float16* p1r1 = pack(L1_r[1], 64, 64, 1);
    _Float16* p1r2 = pack(L1_r[2], 64, 64, 1);
    _Float16* p1r3 = pack(L1_r[3], 64, 512, 1);
    _Float16* p1l10= pack(L1_lin1_0, 128, 128, 1);
    _Float16* p1l11= pack(L1_lin1_1, 128, 128, 1);
    _Float16* p1l20= pack(L1_lin2_0, 256, 256, 1);
    _Float16* p1l21= pack(L1_lin2_1, 256, 128, 1);
    _Float16* p1s0 = pack(L1_skip0, 128, 256, N_SPEC);
    _Float16* p1s1 = pack(L1_skip1, 128, 128, N_SPEC);
    _Float16* p2r0 = pack(L2_r[0], 8, 64, 1);
    _Float16* p2r1 = pack(L2_r[1], 64, 64, 1);
    _Float16* p2r2 = pack(L2_r[2], 64, 64, 1);
    _Float16* p2r3 = pack(L2_r[3], 64, 256, 1);
    _Float16* p2l10= pack(L2_lin1_0, 128, 128, 1);
    _Float16* p2l11= pack(L2_lin1_1, 128, 128, 1);
    _Float16* p2l20= pack(L2_lin2_0, 256, 128, 1);
    _Float16* p2sk = pack(L2_skip0, 128, 128, N_SPEC);

    edge_geom<<<(N_EDGES + 255) / 256, 256, 0, stream>>>(disp, rb, shv);
    onehot_k <<<(N_NODES + 255) / 256, 256, 0, stream>>>(species, s_buf);

    const size_t NP = N_NODES;
    auto scatter = [&](const float* s1, const float* v1, int n0i, int n1i,
                       int n0t, int n1t, int s1ld, int v1ld, int rld,
                       int a0ld, int a1ld) {
        long total = (long)N_EDGES * (n0t + n1t);
        tp_scatter<<<(unsigned)((total + 255) / 256), 256, 0, stream>>>(
            s1, v1, rbuf, shv, senders, receivers, agg0, agg1,
            n0i, n1i, n0t, n1t, s1ld, v1ld, rld, a0ld, a1ld);
    };

    // ---------- layer 0 -----------------------------------------------------
    gemm(32, 4, rb, p0r0, h1,  N_EDGES, 64, 64, 1.0f, 1);
    gemm(64, 4, h1, p0r1, h2,  N_EDGES, 64, 64, 1.0f, 1);
    gemm(64, 4, h2, p0r2, h1,  N_EDGES, 64, 64, 1.0f, 1);
    gemm(64, 1, h1, p0r3, rbuf,N_EDGES, 16, 16, 1.0f, 0);
    gemm(32, 1, s_buf, p0l1, s1b, N_NODES, 16, 32, is5, 0);
    hipMemsetAsync(agg0, 0, NP * 32 * 4, stream);
    hipMemsetAsync(agg1, 0, 3 * NP * 32 * 4, stream);
    scatter(s1b, s1b, 5, 0, 5, 5, 32, 32, 16, 32, 32);
    gemm (32, 8, agg0, p0l20, out0, N_NODES, 256, 256, is5, 0);
    gskip(32, s_buf, p0sk, out0, N_NODES, 256, 256, is5, 0);
    for (int a = 0; a < 3; ++a)
        gemm(32, 8, agg1 + (size_t)a * NP * 32, p0l21,
             out1 + (size_t)a * NP * 128, N_NODES, 128, 128, is5, 0);
    gate_k<<<(N_NODES * HID + 255) / 256, 256, 0, stream>>>(out0, out1, s_buf, v_buf, 256);

    // ---------- layer 1 -----------------------------------------------------
    gemm(32, 4, rb, p1r0, h1,  N_EDGES, 64, 64, 1.0f, 1);
    gemm(64, 4, h1, p1r1, h2,  N_EDGES, 64, 64, 1.0f, 1);
    gemm(64, 4, h2, p1r2, h1,  N_EDGES, 64, 64, 1.0f, 1);
    gemm(64, 8, h1, p1r3, rbuf,N_EDGES, 512, 512, 1.0f, 0);
    gemm(128, 8, s_buf, p1l10, s1b, N_NODES, 128, 128, is128, 0);
    for (int a = 0; a < 3; ++a)
        gemm(128, 8, v_buf + (size_t)a * NP * 128, p1l11,
             v1b + (size_t)a * NP * 128, N_NODES, 128, 128, is128, 0);
    hipMemsetAsync(agg0, 0, NP * 256 * 4, stream);
    hipMemsetAsync(agg1, 0, 3 * NP * 256 * 4, stream);
    scatter(s1b, v1b, 128, 128, 256, 256, 128, 128, 512, 256, 256);
    gemm (256, 8, agg0, p1l20, out0, N_NODES, 256, 256, is256, 0);
    gskip(128, s_buf, p1s0, out0, N_NODES, 256, 256, is128, 0);
    for (int a = 0; a < 3; ++a) {
        gemm (256, 8, agg1 + (size_t)a * NP * 256, p1l21,
              out1 + (size_t)a * NP * 128, N_NODES, 128, 128, is256, 0);
        gskip(128, v_buf + (size_t)a * NP * 128, p1s1,
              out1 + (size_t)a * NP * 128, N_NODES, 128, 128, is128, 0);
    }
    gate_k<<<(N_NODES * HID + 255) / 256, 256, 0, stream>>>(out0, out1, s_buf, v_buf, 256);

    // ---------- layer 2 (last) ----------------------------------------------
    gemm(32, 4, rb, p2r0, h1,  N_EDGES, 64, 64, 1.0f, 1);
    gemm(64, 4, h1, p2r1, h2,  N_EDGES, 64, 64, 1.0f, 1);
    gemm(64, 4, h2, p2r2, h1,  N_EDGES, 64, 64, 1.0f, 1);
    gemm(64, 8, h1, p2r3, rbuf,N_EDGES, 256, 256, 1.0f, 0);
    gemm(128, 8, s_buf, p2l10, s1b, N_NODES, 128, 128, is128, 0);
    for (int a = 0; a < 3; ++a)
        gemm(128, 8, v_buf + (size_t)a * NP * 128, p2l11,
             v1b + (size_t)a * NP * 128, N_NODES, 128, 128, is128, 0);
    hipMemsetAsync(agg0, 0, NP * 256 * 4, stream);
    scatter(s1b, v1b, 128, 128, 256, 0, 128, 128, 256, 256, 256);
    gemm (256, 8, agg0, p2l20, out0, N_NODES, 128, 128, is256, 0);
    gskip(128, s_buf, p2sk, out0, N_NODES, 128, 128, is128, 1 /*silu*/);

    readout_k<<<(N_NODES + 255) / 256, 256, 0, stream>>>(out0, readout, atom_e,
                                                         species, out);
}
